// HetAgg_14027363188816
// MI455X (gfx1250) — matrix operand order repaired and checked by reading, not compiled, not run on hardware
//
#include <hip/hip_runtime.h>
#include <hip/hip_bf16.h>

typedef __attribute__((ext_vector_type(16))) _Float16 v16h;
typedef __attribute__((ext_vector_type(8)))  _Float16 v8h;
typedef __attribute__((ext_vector_type(4)))  _Float16 v4h;
typedef __attribute__((ext_vector_type(8)))  float    v8f;

#define T_ 3
#define L_ 2
#define D_ 128
#define F_ 128
#define N_ 100000
#define B_ 8192
#define S_ 20

#define ROWPAD 136   // halves per padded row: 272 B, 16B-aligned, bank-skewed
#define WAVES 4
#define BLOCK 128

union V16H { v16h v; v8h h[2]; };

// A fragment (16x32 f16): lane holds row M=lane&15; halves at K = kk*32 + 8*half + [0..7]
// and K = kk*32 + 16 + 8*half + [0..7]  (per CDNA5 ISA 7.12.2).
__device__ inline v16h load_a_frag(const _Float16* rowBase, int kk, int half8) {
    V16H u;
    u.h[0] = *(const v8h*)(rowBase + kk * 32 + 8 * half8);
    u.h[1] = *(const v8h*)(rowBase + kk * 32 + 16 + 8 * half8);
    return u.v;
}

// B fragment (32x16 f16): lane holds col N=lane&15; halves K = kk*32 + 8*half + [0..7] and kk*32 + 16 + 8*half + [0..7],
// read from transposed weights WT[N][K] (row-major, padded stride).
__device__ inline v16h load_b_frag(const _Float16* wtRow, int kk, int half8) {
    V16H u;
    const _Float16* p = wtRow + kk * 32 + 8 * half8;
    u.h[0] = *(const v8h*)(p);
    u.h[1] = *(const v8h*)(p + 16);
    return u.v;
}

// Stage a 128x128 f16 weight matrix (row-major) into padded LDS rows using the
// CDNA5 async global->LDS DMA path (ASYNCcnt-tracked, no VGPR round trip).
// Each lane carries its own LDS byte offset (low 32 bits of the generic shared
// pointer = LDS offset per the ISA aperture truncation rule) and its own
// 64-bit global address; 16 B per lane per instruction.
__device__ inline void stage_weights_async(const _Float16* __restrict__ WT,
                                           _Float16* sW, int tid) {
    for (int c = tid; c < (D_ * D_) / 8; c += BLOCK) {      // 8-half chunks
        int row = c >> 4;
        int col = (c & 15) * 8;
        unsigned int lds_off = (unsigned int)(size_t)(sW + row * ROWPAD + col);
        unsigned long long ga = (unsigned long long)(size_t)(WT + c * 8);
        asm volatile("global_load_async_to_lds_b128 %0, %1, off"
                     :: "v"(lds_off), "v"(ga) : "memory");
    }
    asm volatile("s_wait_asynccnt 0" ::: "memory");
}

// ---- Kernel 1: fold weights. MT = (Wp@Wih)^T f16, c = bp@Wih + bih,
//      WhhT f16, WpT0 f16 ----------------------------------------------------
__global__ void prep_kernel(const float* __restrict__ Wp, const float* __restrict__ bp,
                            const float* __restrict__ Wih, const float* __restrict__ Whh,
                            const float* __restrict__ bih,
                            _Float16* __restrict__ MT, _Float16* __restrict__ WpT0,
                            _Float16* __restrict__ WhhT, float* __restrict__ cvec) {
    int gid = blockIdx.x * blockDim.x + threadIdx.x;
    if (gid >= 6 * D_ * F_) return;
    int ci  = gid >> 14;         // combo = l*T + t
    int rem = gid & 16383;
    int d   = rem >> 7;          // out col
    int f   = rem & 127;         // in row
    int t   = ci % T_;
    const float* wp  = Wp  + (size_t)t  * F_ * D_;
    const float* wih = Wih + (size_t)ci * D_ * D_;
    float acc = 0.f;
    for (int k = 0; k < D_; ++k) acc += wp[f * D_ + k] * wih[k * D_ + d];
    MT[(size_t)ci * D_ * F_ + d * F_ + f]   = (_Float16)acc;                         // MT[d][f]
    WhhT[(size_t)ci * D_ * D_ + d * D_ + f] = (_Float16)Whh[(size_t)ci * D_ * D_ + f * D_ + d];
    if (ci == 0) WpT0[d * F_ + f] = (_Float16)Wp[f * D_ + d];
    if (f == 0) {
        float c = 0.f;
        for (int k = 0; k < D_; ++k) c += bp[t * D_ + k] * wih[k * D_ + d];
        cvec[ci * D_ + d] = c + bih[ci * D_ + d];
    }
}

// ---- Kernel 2: gather rows -> f16 LDS tile -> WMMA GEMM vs staged WT -> f32 out
__global__ __launch_bounds__(BLOCK) void gg_kernel(const float* __restrict__ feats,
                                                   const int* __restrict__ idx,
                                                   const _Float16* __restrict__ WT,
                                                   const float* __restrict__ cvec,
                                                   float* __restrict__ out, int nrows) {
    __shared__ __align__(16) _Float16 sW[D_ * ROWPAD];
    __shared__ __align__(16) _Float16 sX[WAVES][16 * ROWPAD];
    int tid = threadIdx.x;
    stage_weights_async(WT, sW, tid);
    __syncthreads();

    int wave = tid >> 5, lane = tid & 31;
    int row0 = (blockIdx.x * WAVES + wave) * 16;
    if (row0 >= nrows) return;

    _Float16* sx = sX[wave];
    // Gather: one float4 (b128) per lane per row; issue all 16 loads first for MLP.
    float4 tmp[16];
#pragma unroll
    for (int r = 0; r < 16; ++r)
        tmp[r] = ((const float4*)(feats + (size_t)idx[row0 + r] * F_))[lane];
#pragma unroll
    for (int r = 0; r < 16; ++r) {
        v4h hv = { (_Float16)tmp[r].x, (_Float16)tmp[r].y,
                   (_Float16)tmp[r].z, (_Float16)tmp[r].w };
        *(v4h*)(sx + r * ROWPAD + lane * 4) = hv;            // ds_store_b64
    }
    int half8 = (lane >> 4) & 1, l15 = lane & 15;
    v16h a[4];
#pragma unroll
    for (int kk = 0; kk < 4; ++kk) a[kk] = load_a_frag(sx + l15 * ROWPAD, kk, half8);
#pragma unroll
    for (int n = 0; n < 8; ++n) {
        v8f acc = {};
#pragma unroll
        for (int kk = 0; kk < 4; ++kk) {
            v16h b = load_b_frag(sW + (n * 16 + l15) * ROWPAD, kk, half8);
            acc = __builtin_amdgcn_wmma_f32_16x16x32_f16(false, a[kk], false, b,
                                                         (short)0, acc, false, false);
        }
        int col = n * 16 + l15;
        float cv = cvec[col];
#pragma unroll
        for (int v = 0; v < 8; ++v)
            out[(size_t)(row0 + v + 8 * half8) * D_ + col] = acc[v] + cv;
    }
}

// ---- Kernel 3: per-16-row-tile RNN recurrence, H resident in wave-private LDS
__global__ __launch_bounds__(BLOCK) void rnn_kernel(const float* __restrict__ U,
                                                    const _Float16* __restrict__ WT,
                                                    const float* __restrict__ bhh,
                                                    float* __restrict__ agg, int Brows) {
    __shared__ __align__(16) _Float16 sW[D_ * ROWPAD];
    __shared__ __align__(16) _Float16 sH[WAVES][16 * ROWPAD];
    int tid = threadIdx.x;
    stage_weights_async(WT, sW, tid);
    __syncthreads();

    int wave = tid >> 5, lane = tid & 31;
    int row0 = (blockIdx.x * WAVES + wave) * 16;
    if (row0 >= Brows) return;

    _Float16* sh = sH[wave];
    for (int i = lane; i < 16 * ROWPAD; i += 32) sh[i] = (_Float16)0.f;

    int half8 = (lane >> 4) & 1, l15 = lane & 15;
    float hsum[8][8];
#pragma unroll
    for (int n = 0; n < 8; ++n)
#pragma unroll
        for (int v = 0; v < 8; ++v) hsum[n][v] = 0.f;

    for (int s = 0; s < S_; ++s) {
        if (s + 1 < S_) {   // prefetch next step's U rows (global_prefetch_b8)
            __builtin_prefetch(&U[((size_t)(row0 + half8 * 8) * S_ + s + 1) * D_ + l15 * 8], 0, 0);
        }
        v16h a[4];
#pragma unroll
        for (int kk = 0; kk < 4; ++kk) a[kk] = load_a_frag(sh + l15 * ROWPAD, kk, half8);
#pragma unroll
        for (int n = 0; n < 8; ++n) {
            v8f acc = {};
#pragma unroll
            for (int kk = 0; kk < 4; ++kk) {
                v16h b = load_b_frag(sW + (n * 16 + l15) * ROWPAD, kk, half8);
                acc = __builtin_amdgcn_wmma_f32_16x16x32_f16(false, a[kk], false, b,
                                                             (short)0, acc, false, false);
            }
            int col = n * 16 + l15;
            float bb = bhh[col];
#pragma unroll
            for (int v = 0; v < 8; ++v) {
                int m = v + 8 * half8;
                float pre = acc[v] + bb + U[((size_t)(row0 + m) * S_ + s) * D_ + col];
                float h = tanhf(pre);
                hsum[n][v] += h;
                sh[m * ROWPAD + col] = (_Float16)h;
            }
        }
    }
    const float inv = 1.0f / (float)S_;
#pragma unroll
    for (int n = 0; n < 8; ++n) {
        int col = n * 16 + l15;
#pragma unroll
        for (int v = 0; v < 8; ++v)
            agg[(size_t)(row0 + v + 8 * half8) * D_ + col] = hsum[n][v] * inv;
    }
}

// ---- Kernel 4: semantic attention (softmax over 4 slots) + leaky ReLU ------
__global__ void attn_kernel(const float* __restrict__ cur, const float* __restrict__ a0,
                            const float* __restrict__ a1, const float* __restrict__ a2,
                            const float* __restrict__ semw, float* __restrict__ out, int nb) {
    int b = blockIdx.x * blockDim.x + threadIdx.x;
    if (b >= nb) return;
    const float* rc = cur + (size_t)b * D_;
    const float* r1 = a0 + (size_t)b * D_;
    const float* r2 = a1 + (size_t)b * D_;
    const float* r3 = a2 + (size_t)b * D_;
    float base = 0.f, s0 = 0.f, s1 = 0.f, s2 = 0.f, s3 = 0.f;
    for (int d = 0; d < D_; ++d) {
        float wl = semw[d], wh = semw[D_ + d], c = rc[d];
        base += wl * c;
        s0 += wh * c; s1 += wh * r1[d]; s2 += wh * r2[d]; s3 += wh * r3[d];
    }
    float l0 = base + s0, l1 = base + s1, l2 = base + s2, l3 = base + s3;
    float mx = fmaxf(fmaxf(l0, l1), fmaxf(l2, l3));
    float e0 = __expf(l0 - mx), e1 = __expf(l1 - mx), e2 = __expf(l2 - mx), e3 = __expf(l3 - mx);
    float inv = 1.f / (e0 + e1 + e2 + e3);
    e0 *= inv; e1 *= inv; e2 *= inv; e3 *= inv;
    for (int d = 0; d < D_; ++d) {
        float v = e0 * rc[d] + e1 * r1[d] + e2 * r2[d] + e3 * r3[d];
        out[(size_t)b * D_ + d] = v > 0.f ? v : 0.01f * v;
    }
}

extern "C" void kernel_launch(void* const* d_in, const int* in_sizes, int n_in,
                              void* d_out, int out_size, void* d_ws, size_t ws_size,
                              hipStream_t stream) {
    const float* feats = (const float*)d_in[0];
    const float* Wp    = (const float*)d_in[1];
    const float* bp    = (const float*)d_in[2];
    const float* Wih   = (const float*)d_in[3];
    const float* Whh   = (const float*)d_in[4];
    const float* bih   = (const float*)d_in[5];
    const float* bhh   = (const float*)d_in[6];
    const float* semw  = (const float*)d_in[7];
    const int* self_idx  = (const int*)d_in[8];
    const int* neigh_idx = (const int*)d_in[9];

    char* ws = (char*)d_ws;
    size_t o = 0;
    _Float16* MT   = (_Float16*)(ws + o); o += (size_t)6 * D_ * F_ * sizeof(_Float16);
    _Float16* WpT0 = (_Float16*)(ws + o); o += (size_t)D_ * F_ * sizeof(_Float16);
    _Float16* WhhT = (_Float16*)(ws + o); o += (size_t)6 * D_ * D_ * sizeof(_Float16);
    float* cvec    = (float*)(ws + o);    o += (size_t)6 * D_ * sizeof(float);
    o = (o + 255) & ~(size_t)255;
    float* cur0 = (float*)(ws + o); o += (size_t)B_ * D_ * sizeof(float);
    float* cur1 = (float*)(ws + o); o += (size_t)B_ * D_ * sizeof(float);
    float* agg  = (float*)(ws + o); o += (size_t)6 * B_ * D_ * sizeof(float);
    float* U    = (float*)(ws + o); o += (size_t)B_ * S_ * D_ * sizeof(float);

    prep_kernel<<<(6 * D_ * F_ + 255) / 256, 256, 0, stream>>>(Wp, bp, Wih, Whh, bih,
                                                               MT, WpT0, WhhT, cvec);
    // self projection: cur0 = feats[0][self_idx] @ Wp[0] + bp[0]
    gg_kernel<<<(B_ / 16) / WAVES, BLOCK, 0, stream>>>(feats, self_idx, WpT0, bp, cur0, B_);

    for (int ci = 0; ci < 6; ++ci) {
        int t = ci % T_;
        gg_kernel<<<(B_ * S_ / 16) / WAVES, BLOCK, 0, stream>>>(
            feats + (size_t)t * N_ * F_, neigh_idx + (size_t)ci * B_ * S_,
            MT + (size_t)ci * D_ * F_, cvec + ci * D_, U, B_ * S_);
        rnn_kernel<<<(B_ / 16) / WAVES, BLOCK, 0, stream>>>(
            U, WhhT + (size_t)ci * D_ * D_, bhh + ci * D_, agg + (size_t)ci * B_ * D_, B_);
    }

    attn_kernel<<<(B_ + 255) / 256, 256, 0, stream>>>(
        cur0, agg, agg + (size_t)B_ * D_, agg + (size_t)2 * B_ * D_, semw, cur1, B_);
    attn_kernel<<<(B_ + 255) / 256, 256, 0, stream>>>(
        cur1, agg + (size_t)3 * B_ * D_, agg + (size_t)4 * B_ * D_, agg + (size_t)5 * B_ * D_,
        semw + 2 * D_, (float*)d_out, B_);
    (void)in_sizes; (void)n_in; (void)out_size; (void)ws_size;
}
